// MixedSparseGatedMLP_14121852469624
// MI455X (gfx1250) — matrix operand order repaired and checked by reading, not compiled, not run on hardware
//
#include <hip/hip_runtime.h>

// ---------------------------------------------------------------------------
// QLoRA gated MLP for MI455X (gfx1250, wave32, WMMA + async-to-LDS).
//   M = B*S = 4096, DM = 4096, DFF = 11008, NF4 block = 64, rank = 16.
// Compute-bound (~1.11 TFLOP vs ~0.7 GB traffic), so:
//   1) dequant+LoRA-fold weights once into fp16 W'[out][in] (= [N][K])
//   2) x1 fp32 -> fp16
//   3) fused gate/up GEMM (v_wmma_f32_16x16x32_f16) + relu*mul -> x3 fp16
//   4) down GEMM -> fp32 out
// GEMMs use GLOBAL_LOAD_ASYNC_TO_LDS_B128 with a 3-stage LDS pipeline
// (ASYNCcnt-gated), K-step 64, bank-conflict-free padded LDS tiles.
// ---------------------------------------------------------------------------

#define DM    4096
#define DFF   11008
#define MTOT  4096   // B*S
#define KSTEP 64
#define NSTAGE 3
#define LDSTRIDE 72  // halves per LDS tile row (64 data + 8 pad -> 144B rows)

typedef _Float16 v16h __attribute__((ext_vector_type(16)));
typedef _Float16 v8h  __attribute__((ext_vector_type(8)));
typedef float    v8f  __attribute__((ext_vector_type(8)));

__constant__ float NF4_LUT[16] = {
    -1.0f, -0.6961928009986877f, -0.5250730514526367f, -0.39491748809814453f,
    -0.28444138169288635f, -0.18477343022823334f, -0.09105003625154495f, 0.0f,
    0.07958029955625534f, 0.16093020141124725f, 0.2461123019456863f,
    0.33791524171829224f, 0.44070982933044434f, 0.5626170039176941f,
    0.7229568362236023f, 1.0f};

// ---------------------------------------------------------------------------
// Async memory->LDS copy (16B per lane). ASYNCcnt-tracked; flat->LDS address
// is the low 32 bits of the generic pointer (ISA 10.2 aperture truncation).
// ---------------------------------------------------------------------------
__device__ __forceinline__ void async_copy_b128(_Float16* lds_dst,
                                                const _Float16* gsrc) {
  unsigned lds_off = (unsigned)(unsigned long long)lds_dst;
  asm volatile("global_load_async_to_lds_b128 %0, %1, off"
               :
               : "v"(lds_off), "v"(gsrc)
               : "memory");
}

// ---------------------------------------------------------------------------
// Kernel 1: fp32 -> fp16 conversion of activations (8 elements / thread)
// ---------------------------------------------------------------------------
__global__ __launch_bounds__(256) void f32_to_f16_kernel(
    const float* __restrict__ in, _Float16* __restrict__ out, long n) {
  long i = (((long)blockIdx.x * blockDim.x) + threadIdx.x) * 8;
  if (i + 8 > n) return;
  v8h o;
#pragma unroll
  for (int j = 0; j < 8; ++j) o[j] = (_Float16)in[i + j];
  *(v8h*)(out + i) = o;
}

// ---------------------------------------------------------------------------
// Kernel 2: NF4 dequant + LoRA fold.
//   out[n][k] = NF4[codes[n][k]] * absmax[n][k/64] + sum_r la[k][r]*lb[r][n]
// ---------------------------------------------------------------------------
__global__ __launch_bounds__(256) void dequant_lora_kernel(
    const int* __restrict__ codes, const float* __restrict__ absmax,
    const float* __restrict__ la, const float* __restrict__ lb,
    _Float16* __restrict__ out, int N, int K) {
  long tid = (long)blockIdx.x * blockDim.x + threadIdx.x;
  int kchunks = K >> 3;
  if (tid >= (long)N * kchunks) return;
  int n  = (int)(tid / kchunks);
  int kc = (int)(tid % kchunks);
  int k0 = kc << 3;

  float bv[16];
#pragma unroll
  for (int r = 0; r < 16; ++r) bv[r] = lb[(long)r * N + n];

  float am = absmax[(long)n * (K >> 6) + (k0 >> 6)];
  const int* cp = codes + (long)n * K + k0;

  v8h o;
#pragma unroll
  for (int j = 0; j < 8; ++j) {
    float w = NF4_LUT[cp[j] & 15] * am;
    const float* ar = la + (long)(k0 + j) * 16;
    float acc = 0.f;
#pragma unroll
    for (int r = 0; r < 16; ++r) acc += ar[r] * bv[r];
    o[j] = (_Float16)(w + acc);
  }
  *(v8h*)(out + (long)n * K + k0) = o;
}

// ---------------------------------------------------------------------------
// WMMA fragment helpers (LDS -> VGPR), per CDNA5 ISA 7.12.2 layouts.
//   A 16x32 f16: lane L = row M=L%16; lanes 0-15 K{0-7,16-23}, 16-31 K{8-15,24-31}
//   B 32x16 f16: lane L = col N=L%16; lanes 0-15 K{0-7,16-23}, 16-31 K{8-15,24-31}
// Tiles row-major, stride LDSTRIDE halves (padded: 16 rows -> 16 distinct banks).
// ---------------------------------------------------------------------------
__device__ __forceinline__ v16h lds_a_frag(const _Float16* tile, int row,
                                           int ks, int l16, int sel) {
  const _Float16* p = tile + (row + l16) * LDSTRIDE + ks * 32;
  v8h p0 = *(const v8h*)(p + sel * 8);
  v8h p1 = *(const v8h*)(p + 16 + sel * 8);
  v16h r;
#pragma unroll
  for (int i = 0; i < 8; ++i) { r[i] = p0[i]; r[i + 8] = p1[i]; }
  return r;
}

__device__ __forceinline__ v16h lds_b_frag(const _Float16* tile, int col,
                                           int ks, int l16, int sel) {
  const _Float16* p = tile + (col + l16) * LDSTRIDE + ks * 32;
  v8h p0 = *(const v8h*)(p + sel * 8);
  v8h p1 = *(const v8h*)(p + 16 + sel * 8);
  v16h r;
#pragma unroll
  for (int i = 0; i < 8; ++i) { r[i] = p0[i]; r[i + 8] = p1[i]; }
  return r;
}

// ---------------------------------------------------------------------------
// Kernel 3: fused gate/up GEMM + relu*mul epilogue.
//   X [M][DM] f16, Wg/Wu [DFF][DM] f16 ([N][K]), X3 [M][DFF] f16.
//   Block 128(M) x 64(N), K-step 64, 8 waves = 4(M) x 2(N), wave 32x32 per
//   weight matrix. 3-stage async LDS pipeline; L = 8 async b128 per thread
//   per stage -> asynccnt gates 16 / 8 / 0.
// ---------------------------------------------------------------------------
__global__ __launch_bounds__(256) void gemm_gate_up_kernel(
    const _Float16* __restrict__ X, const _Float16* __restrict__ Wg,
    const _Float16* __restrict__ Wu, _Float16* __restrict__ X3) {
  __shared__ _Float16 As[NSTAGE][128 * LDSTRIDE];
  __shared__ _Float16 Bgs[NSTAGE][64 * LDSTRIDE];
  __shared__ _Float16 Bus[NSTAGE][64 * LDSTRIDE];

  const int tid  = threadIdx.x;
  const int lane = tid & 31;
  const int wv   = tid >> 5;
  const int wm   = (wv & 3) * 32;
  const int wn   = (wv >> 2) * 32;
  const int l16  = lane & 15;
  const int sel  = lane >> 4;
  const int m0   = blockIdx.y * 128;
  const int n0   = blockIdx.x * 64;

  // staging geometry: 8-half chunk ca of row ra (0..31); 4 row-passes for A
  const int ca = (tid & 7) * 8;
  const int ra = tid >> 3;

  const _Float16* gA = X + (long)(m0 + ra) * DM + ca;
  const _Float16* gG = Wg + (long)(n0 + ra) * DM + ca;
  const _Float16* gU = Wu + (long)(n0 + ra) * DM + ca;

  v8f accg[2][2], accu[2][2];
  v8f z;
#pragma unroll
  for (int i = 0; i < 8; ++i) z[i] = 0.f;
#pragma unroll
  for (int i = 0; i < 2; ++i)
#pragma unroll
    for (int j = 0; j < 2; ++j) { accg[i][j] = z; accu[i][j] = z; }

  const int NK = DM / KSTEP;  // 64

  auto issue_stage = [&](int s, int kt) {
    const long k0 = (long)kt * KSTEP;
#pragma unroll
    for (int i = 0; i < 4; ++i)
      async_copy_b128(&As[s][(ra + i * 32) * LDSTRIDE + ca],
                      gA + k0 + (long)i * 32 * DM);
#pragma unroll
    for (int i = 0; i < 2; ++i) {
      async_copy_b128(&Bgs[s][(ra + i * 32) * LDSTRIDE + ca],
                      gG + k0 + (long)i * 32 * DM);
      async_copy_b128(&Bus[s][(ra + i * 32) * LDSTRIDE + ca],
                      gU + k0 + (long)i * 32 * DM);
    }
  };

#pragma unroll
  for (int s = 0; s < NSTAGE; ++s) issue_stage(s, s);

  for (int kt = 0; kt < NK; ++kt) {
    const int buf = kt % NSTAGE;
    // stages kt .. min(NK-1, kt+2) are in flight; require stage kt complete
    const int ahead = (kt + 2 <= NK - 1) ? 2 : ((kt + 1 <= NK - 1) ? 1 : 0);
    if (ahead == 2)      asm volatile("s_wait_asynccnt 16" ::: "memory");
    else if (ahead == 1) asm volatile("s_wait_asynccnt 8" ::: "memory");
    else                 asm volatile("s_wait_asynccnt 0" ::: "memory");
    __syncthreads();

#pragma unroll
    for (int ks = 0; ks < 2; ++ks) {
      v16h af[2], bg[2], bu[2];
#pragma unroll
      for (int fm = 0; fm < 2; ++fm)
        af[fm] = lds_a_frag(&As[buf][0], wm + fm * 16, ks, l16, sel);
#pragma unroll
      for (int fn = 0; fn < 2; ++fn) {
        bg[fn] = lds_b_frag(&Bgs[buf][0], wn + fn * 16, ks, l16, sel);
        bu[fn] = lds_b_frag(&Bus[buf][0], wn + fn * 16, ks, l16, sel);
      }
#pragma unroll
      for (int fm = 0; fm < 2; ++fm)
#pragma unroll
        for (int fn = 0; fn < 2; ++fn) {
          accg[fm][fn] = __builtin_amdgcn_wmma_f32_16x16x32_f16(
              false, af[fm], false, bg[fn], (short)0, accg[fm][fn], false,
              false);
          accu[fm][fn] = __builtin_amdgcn_wmma_f32_16x16x32_f16(
              false, af[fm], false, bu[fn], (short)0, accu[fm][fn], false,
              false);
        }
    }

    __syncthreads();  // all waves done reading buf before refilling it
    if (kt + NSTAGE < NK) issue_stage(buf, kt + NSTAGE);
  }

  // epilogue: x3 = relu(gate) * up. C frag: VGPR v, lane l:
  //   row = v + (l>=16 ? 8 : 0), col = l%16
#pragma unroll
  for (int fm = 0; fm < 2; ++fm) {
    const int row = m0 + wm + fm * 16 + sel * 8;
#pragma unroll
    for (int fn = 0; fn < 2; ++fn) {
      const int col = n0 + wn + fn * 16 + l16;
#pragma unroll
      for (int v = 0; v < 8; ++v) {
        float g = accg[fm][fn][v];
        float u = accu[fm][fn][v];
        float x = (g > 0.f ? g : 0.f) * u;
        X3[(long)(row + v) * DFF + col] = (_Float16)x;
      }
    }
  }
}

// ---------------------------------------------------------------------------
// Kernel 4: down GEMM. X3 [M][DFF] f16, Wd [DM][DFF] f16 ([N][K]),
//   Y [M][DM] f32. Same tiling; L = 6 async b128 per thread per stage
//   -> asynccnt gates 12 / 6 / 0. NK = 172.
// ---------------------------------------------------------------------------
__global__ __launch_bounds__(256) void gemm_down_kernel(
    const _Float16* __restrict__ X3, const _Float16* __restrict__ Wd,
    float* __restrict__ Y) {
  __shared__ _Float16 As[NSTAGE][128 * LDSTRIDE];
  __shared__ _Float16 Bs[NSTAGE][64 * LDSTRIDE];

  const int tid  = threadIdx.x;
  const int lane = tid & 31;
  const int wv   = tid >> 5;
  const int wm   = (wv & 3) * 32;
  const int wn   = (wv >> 2) * 32;
  const int l16  = lane & 15;
  const int sel  = lane >> 4;
  const int m0   = blockIdx.y * 128;
  const int n0   = blockIdx.x * 64;

  const int ca = (tid & 7) * 8;
  const int ra = tid >> 3;

  const _Float16* gA = X3 + (long)(m0 + ra) * DFF + ca;
  const _Float16* gB = Wd + (long)(n0 + ra) * DFF + ca;

  v8f acc[2][2];
  v8f z;
#pragma unroll
  for (int i = 0; i < 8; ++i) z[i] = 0.f;
#pragma unroll
  for (int i = 0; i < 2; ++i)
#pragma unroll
    for (int j = 0; j < 2; ++j) acc[i][j] = z;

  const int NK = DFF / KSTEP;  // 172

  auto issue_stage = [&](int s, int kt) {
    const long k0 = (long)kt * KSTEP;
#pragma unroll
    for (int i = 0; i < 4; ++i)
      async_copy_b128(&As[s][(ra + i * 32) * LDSTRIDE + ca],
                      gA + k0 + (long)i * 32 * DFF);
#pragma unroll
    for (int i = 0; i < 2; ++i)
      async_copy_b128(&Bs[s][(ra + i * 32) * LDSTRIDE + ca],
                      gB + k0 + (long)i * 32 * DFF);
  };

#pragma unroll
  for (int s = 0; s < NSTAGE; ++s) issue_stage(s, s);

  for (int kt = 0; kt < NK; ++kt) {
    const int buf = kt % NSTAGE;
    const int ahead = (kt + 2 <= NK - 1) ? 2 : ((kt + 1 <= NK - 1) ? 1 : 0);
    if (ahead == 2)      asm volatile("s_wait_asynccnt 12" ::: "memory");
    else if (ahead == 1) asm volatile("s_wait_asynccnt 6" ::: "memory");
    else                 asm volatile("s_wait_asynccnt 0" ::: "memory");
    __syncthreads();

#pragma unroll
    for (int ks = 0; ks < 2; ++ks) {
      v16h af[2], bf[2];
#pragma unroll
      for (int fm = 0; fm < 2; ++fm)
        af[fm] = lds_a_frag(&As[buf][0], wm + fm * 16, ks, l16, sel);
#pragma unroll
      for (int fn = 0; fn < 2; ++fn)
        bf[fn] = lds_b_frag(&Bs[buf][0], wn + fn * 16, ks, l16, sel);
#pragma unroll
      for (int fm = 0; fm < 2; ++fm)
#pragma unroll
        for (int fn = 0; fn < 2; ++fn)
          acc[fm][fn] = __builtin_amdgcn_wmma_f32_16x16x32_f16(
              false, af[fm], false, bf[fn], (short)0, acc[fm][fn], false,
              false);
    }

    __syncthreads();
    if (kt + NSTAGE < NK) issue_stage(buf, kt + NSTAGE);
  }

#pragma unroll
  for (int fm = 0; fm < 2; ++fm) {
    const int row = m0 + wm + fm * 16 + sel * 8;
#pragma unroll
    for (int fn = 0; fn < 2; ++fn) {
      const int col = n0 + wn + fn * 16 + l16;
#pragma unroll
      for (int v = 0; v < 8; ++v)
        Y[(long)(row + v) * DM + col] = acc[fm][fn][v];
    }
  }
}

// ---------------------------------------------------------------------------
// Host launcher
// ---------------------------------------------------------------------------
extern "C" void kernel_launch(void* const* d_in, const int* in_sizes, int n_in,
                              void* d_out, int out_size, void* d_ws,
                              size_t ws_size, hipStream_t stream) {
  (void)in_sizes; (void)n_in; (void)out_size; (void)ws_size;

  const float* x1       = (const float*)d_in[0];
  const int*   g_codes  = (const int*)d_in[1];
  const float* g_absmax = (const float*)d_in[2];
  const float* g_la     = (const float*)d_in[3];
  const float* g_lb     = (const float*)d_in[4];
  const int*   u_codes  = (const int*)d_in[5];
  const float* u_absmax = (const float*)d_in[6];
  const float* u_la     = (const float*)d_in[7];
  const float* u_lb     = (const float*)d_in[8];
  const int*   d_codes  = (const int*)d_in[9];
  const float* d_absmax = (const float*)d_in[10];
  const float* d_la     = (const float*)d_in[11];
  const float* d_lb     = (const float*)d_in[12];
  float* out = (float*)d_out;

  // workspace layout (bytes), total ~394 MB
  const size_t SZ_X1H = (size_t)MTOT * DM * 2;
  const size_t SZ_W   = (size_t)DFF * DM * 2;
  char* ws = (char*)d_ws;
  _Float16* x1h = (_Float16*)(ws);
  _Float16* wg  = (_Float16*)(ws + SZ_X1H);
  _Float16* wu  = (_Float16*)(ws + SZ_X1H + SZ_W);
  _Float16* wd  = (_Float16*)(ws + SZ_X1H + 2 * SZ_W);
  _Float16* x3h = (_Float16*)(ws + SZ_X1H + 3 * SZ_W);

  {  // 1) x1 -> fp16
    long n = (long)MTOT * DM;
    int blocks = (int)((n / 8 + 255) / 256);
    f32_to_f16_kernel<<<blocks, 256, 0, stream>>>(x1, x1h, n);
  }

  {  // 2) dequant + LoRA fold
    long tg = (long)DFF * (DM / 8);
    int blocks = (int)((tg + 255) / 256);
    dequant_lora_kernel<<<blocks, 256, 0, stream>>>(g_codes, g_absmax, g_la,
                                                    g_lb, wg, DFF, DM);
    dequant_lora_kernel<<<blocks, 256, 0, stream>>>(u_codes, u_absmax, u_la,
                                                    u_lb, wu, DFF, DM);
    long td = (long)DM * (DFF / 8);
    int blocksd = (int)((td + 255) / 256);
    dequant_lora_kernel<<<blocksd, 256, 0, stream>>>(d_codes, d_absmax, d_la,
                                                     d_lb, wd, DM, DFF);
  }

  // 3) fused gate/up GEMM + relu*mul
  gemm_gate_up_kernel<<<dim3(DFF / 64, MTOT / 128), 256, 0, stream>>>(
      x1h, wg, wu, x3h);

  // 4) down GEMM -> fp32 output
  gemm_down_kernel<<<dim3(DM / 64, MTOT / 128), 256, 0, stream>>>(x3h, wd, out);
}
